// DiffTransformerLayer_5531917877771
// MI455X (gfx1250) — matrix operand order repaired and checked by reading, not compiled, not run on hardware
//
#include <hip/hip_runtime.h>
#include <hip/hip_bf16.h>
#include <math.h>

// ---------------------------------------------------------------------------
// Problem constants (DiffTransformer layer): B=2, S=2048, D=1024, H=8, HD=64,
// 2H=16 softmax heads, FFN=2048.  LAMBDA_INIT = 0.8 - 0.6*exp(-0.3*11).
// ---------------------------------------------------------------------------
#define BSZ   2
#define SEQ   2048
#define DIM   1024
#define NH    8            // diff heads
#define HD2   128          // 2*HD, v head dim
#define HDQ   64           // per-softmax-head dim
#define FFN   2048
#define MROWS (BSZ*SEQ)    // 4096
#define LAMBDA_INIT 0.77787010f

typedef __attribute__((ext_vector_type(16))) _Float16 v16h;
typedef __attribute__((ext_vector_type(8)))  float    v8f;
typedef __attribute__((ext_vector_type(4)))  unsigned int u32x4;
typedef __attribute__((ext_vector_type(4)))  int          i32x4;

union Frag16 { v16h h; u32x4 u[2]; };

// ---------------------------------------------------------------------------
// CDNA5 async global->LDS copy (ASYNCcnt path), with synchronous fallback.
// Builtin expects int4 pointers: addrspace(1) for global, addrspace(3) for
// LDS.  LDS pointer built from the flat shared address's low 32 bits (ISA
// aperture mapping: flat-LDS addr[31:0] is the LDS byte address).
// ---------------------------------------------------------------------------
#if __has_builtin(__builtin_amdgcn_global_load_async_to_lds_b128)
#define HAVE_ASYNC_LDS 1
#else
#define HAVE_ASYNC_LDS 0
#endif

typedef __attribute__((address_space(1))) i32x4 gas_i32x4;
typedef __attribute__((address_space(3))) i32x4 las_i32x4;

__device__ __forceinline__ void copy_b128_to_lds(const _Float16* g, _Float16* l) {
#if HAVE_ASYNC_LDS
  gas_i32x4* gp = (gas_i32x4*)(unsigned long long)(const void*)g;
  las_i32x4* lp = (las_i32x4*)(unsigned int)(unsigned long long)(void*)l;
  __builtin_amdgcn_global_load_async_to_lds_b128(gp, lp, 0, 0);
#else
  *(u32x4*)l = *(const u32x4*)g;
#endif
}

__device__ __forceinline__ void wait_async_lds() {
#if HAVE_ASYNC_LDS
#if __has_builtin(__builtin_amdgcn_s_wait_asynccnt)
  __builtin_amdgcn_s_wait_asynccnt(0);
#else
  asm volatile("s_wait_asynccnt 0" ::: "memory");
#endif
#endif
}

// ---------------------------------------------------------------------------
// wave32 reductions
// ---------------------------------------------------------------------------
__device__ __forceinline__ float waveSum(float v) {
#pragma unroll
  for (int o = 16; o > 0; o >>= 1) v += __shfl_xor(v, o, 32);
  return v;
}
__device__ __forceinline__ float waveMax(float v) {
#pragma unroll
  for (int o = 16; o > 0; o >>= 1) v = fmaxf(v, __shfl_xor(v, o, 32));
  return v;
}

// ---------------------------------------------------------------------------
// WMMA GEMM:  C[M,N] = A[M,K] (f16, row-major, lda)  x  BT[N,K] (f16, ldb)
// Block: 256 threads = 8 waves. Block tile 128x64, K-step 32, double-buffered
// LDS staging filled with GLOBAL_LOAD_ASYNC_TO_LDS_B128.
// Each wave: 16 rows x 64 cols (4 wmma tiles). Grid: (N/64, M/128).
// ---------------------------------------------------------------------------
enum { EPI_F32 = 0, EPI_F16 = 1, EPI_QK = 2, EPI_V = 3,
       EPI_SCORES = 4, EPI_RESID = 5, EPI_SILU = 6 };

#define AST 40   // padded LDS row stride in halves (80B = 20 banks)

__device__ __forceinline__ void stage_tiles(
    const _Float16* __restrict__ A, const _Float16* __restrict__ BT,
    int lda, int ldb, int m0blk, int n0blk, int k,
    _Float16* sA, _Float16* sB, int tid)
{
  // A tile 128x32 halves = 512 b128 chunks; 2 per thread.
  const int r  = tid >> 2;
  const int c8 = (tid & 3) * 8;
  copy_b128_to_lds(A + (size_t)(m0blk + r)      * lda + k + c8, sA + r        * AST + c8);
  copy_b128_to_lds(A + (size_t)(m0blk + r + 64) * lda + k + c8, sA + (r + 64) * AST + c8);
  // B tile 64x32 halves = 256 b128 chunks; 1 per thread.
  copy_b128_to_lds(BT + (size_t)(n0blk + r) * ldb + k + c8, sB + r * AST + c8);
}

template <int MODE>
__global__ __launch_bounds__(256) void gemm_wmma(
    const _Float16* __restrict__ A, const _Float16* __restrict__ BT,
    float* __restrict__ Cf, _Float16* __restrict__ Ch,
    const float* __restrict__ aux,
    int K, int lda, int ldb, int ldc)
{
  __shared__ _Float16 shA[2][128 * AST];
  __shared__ _Float16 shB[2][64 * AST];

  const int tid    = threadIdx.x;
  const int lane   = tid & 31;
  const int wave   = tid >> 5;
  const int l15    = lane & 15;
  const int half16 = lane >> 4;                  // 0: lanes 0-15, 1: lanes 16-31
  const int m0blk  = blockIdx.y * 128;
  const int n0blk  = blockIdx.x * 64;
  const int m0     = m0blk + wave * 16;

  v8f acc[4] = {};

  stage_tiles(A, BT, lda, ldb, m0blk, n0blk, 0, shA[0], shB[0], tid);
  wait_async_lds();
  __syncthreads();

  int cur = 0;
  for (int k = 0; k < K; k += 32) {
    if (k + 32 < K)
      stage_tiles(A, BT, lda, ldb, m0blk, n0blk, k + 32, shA[cur ^ 1], shB[cur ^ 1], tid);

    const _Float16* sA = shA[cur];
    const _Float16* sB = shB[cur];

    // A fragment (ISA 16-bit A 16x32 layout): lane-half selects K base {0,8};
    // two contiguous 8-half chunks at +0 and +16.
    Frag16 af;
    const _Float16* ap = sA + (wave * 16 + l15) * AST + half16 * 8;
    af.u[0] = *(const u32x4*)(ap);
    af.u[1] = *(const u32x4*)(ap + 16);
#pragma unroll
    for (int t = 0; t < 4; ++t) {
      // B fragment: lane holds column n = t*16+l15; lane-half selects K base
      // {0,8}; two contiguous 8-half chunks at +0 and +16 (matches A).
      const _Float16* bp = sB + (t * 16 + l15) * AST + half16 * 8;
      Frag16 bf;
      bf.u[0] = *(const u32x4*)(bp);
      bf.u[1] = *(const u32x4*)(bp + 16);
      acc[t] = __builtin_amdgcn_wmma_f32_16x16x32_f16(
          false, af.h, false, bf.h, (short)0, acc[t], false, false);
    }

    wait_async_lds();
    __syncthreads();
    cur ^= 1;
  }

  // Epilogue: C VGPR r holds M = r (lanes 0-15) / 8+r (lanes 16-31), N = l15.
#pragma unroll
  for (int t = 0; t < 4; ++t) {
#pragma unroll
    for (int r = 0; r < 8; ++r) {
      const int m = m0 + half16 * 8 + r;
      const int n = n0blk + t * 16 + l15;
      float val = acc[t][r];
      if constexpr (MODE == EPI_F32) {
        Cf[(size_t)m * ldc + n] = val;
      } else if constexpr (MODE == EPI_F16) {
        Ch[(size_t)m * ldc + n] = (_Float16)val;
      } else if constexpr (MODE == EPI_QK) {
        // m = b*SEQ + s in [0,4096); n = head*64 + d; scatter head-major.
        const int b = m >> 11, s = m & (SEQ - 1);
        const int hh = n >> 6, d = n & 63;
        Ch[(((size_t)(b * 16 + hh)) * SEQ + s) * HDQ + d] = (_Float16)val;
      } else if constexpr (MODE == EPI_V) {
        // scatter v transposed head-major: vT[b][vh][d][s]
        const int b = m >> 11, s = m & (SEQ - 1);
        const int vh = n >> 7, d = n & 127;
        Ch[(((size_t)(b * NH + vh)) * HD2 + d) * SEQ + s] = (_Float16)val;
      } else if constexpr (MODE == EPI_SCORES) {
        val *= 0.125f;                      // HD^-0.5
        if (n > m) val = -1.0e9f;           // causal mask
        Cf[(size_t)m * ldc + n] = val;
      } else if constexpr (MODE == EPI_RESID) {
        Cf[(size_t)m * ldc + n] = val + aux[(size_t)m * ldc + n];
      } else if constexpr (MODE == EPI_SILU) {
        const float g = aux[(size_t)m * ldc + n];
        const float sg = g / (1.0f + expf(-g));
        Ch[(size_t)m * ldc + n] = (_Float16)(sg * val);
      }
    }
  }
}

// ---------------------------------------------------------------------------
// Cast + transpose fp32 [K,N] -> f16 [N,K]
// ---------------------------------------------------------------------------
__global__ __launch_bounds__(256) void transpose_cast_kernel(
    const float* __restrict__ in, _Float16* __restrict__ out, int K, int N)
{
  __shared__ float tile[32][33];
  const int n0 = blockIdx.x * 32, k0 = blockIdx.y * 32;
  const int tx = threadIdx.x & 31, ty = threadIdx.x >> 5;   // 32 x 8
#pragma unroll
  for (int i = ty; i < 32; i += 8)
    tile[i][tx] = in[(size_t)(k0 + i) * N + n0 + tx];
  __syncthreads();
#pragma unroll
  for (int i = ty; i < 32; i += 8)
    out[(size_t)(n0 + i) * K + k0 + tx] = (_Float16)tile[tx][i];
}

// ---------------------------------------------------------------------------
// LayerNorm (row of 1024) -> f16
// ---------------------------------------------------------------------------
__global__ __launch_bounds__(256) void layernorm_kernel(
    const float* __restrict__ x, const float* __restrict__ w,
    const float* __restrict__ b, _Float16* __restrict__ out)
{
  const int row = blockIdx.x;
  const float* xr = x + (size_t)row * DIM;
  float vals[4], s = 0.f, q = 0.f;
#pragma unroll
  for (int i = 0; i < 4; ++i) {
    const int c = i * 256 + threadIdx.x;
    const float v = xr[c];
    vals[i] = v; s += v; q += v * v;
  }
  s = waveSum(s); q = waveSum(q);
  __shared__ float shS[8], shQ[8];
  if ((threadIdx.x & 31) == 0) { shS[threadIdx.x >> 5] = s; shQ[threadIdx.x >> 5] = q; }
  __syncthreads();
  float ts = 0.f, tq = 0.f;
#pragma unroll
  for (int i = 0; i < 8; ++i) { ts += shS[i]; tq += shQ[i]; }
  const float mean = ts * (1.0f / DIM);
  const float var  = tq * (1.0f / DIM) - mean * mean;
  const float inv  = rsqrtf(var + 1e-5f);
#pragma unroll
  for (int i = 0; i < 4; ++i) {
    const int c = i * 256 + threadIdx.x;
    out[(size_t)row * DIM + c] = (_Float16)((vals[i] - mean) * inv * w[c] + b[c]);
  }
}

// ---------------------------------------------------------------------------
// lambda = exp(sum lq1*lk1) - exp(sum lq2*lk2) + LAMBDA_INIT   (HD=64)
// ---------------------------------------------------------------------------
__global__ void lam_kernel(const float* __restrict__ lq1, const float* __restrict__ lk1,
                           const float* __restrict__ lq2, const float* __restrict__ lk2,
                           float* __restrict__ lam)
{
  const int t = threadIdx.x;   // 32 threads
  float a = lq1[t] * lk1[t] + lq1[t + 32] * lk1[t + 32];
  float b = lq2[t] * lk2[t] + lq2[t + 32] * lk2[t + 32];
  a = waveSum(a); b = waveSum(b);
  if (t == 0) *lam = expf(a) - expf(b) + LAMBDA_INIT;
}

// ---------------------------------------------------------------------------
// Fused dual softmax + diff: a = softmax(s1) - lam * softmax(s2), f16 out.
// One block per row (2048 cols, 8 per thread).
// ---------------------------------------------------------------------------
__global__ __launch_bounds__(256) void softdiff_kernel(
    const float* __restrict__ s1, const float* __restrict__ s2,
    const float* __restrict__ lamp, _Float16* __restrict__ a)
{
  const int row = blockIdx.x;
  const float* p1 = s1 + (size_t)row * SEQ;
  const float* p2 = s2 + (size_t)row * SEQ;
  float r1[8], r2[8], m1 = -3.0e38f, m2 = -3.0e38f;
#pragma unroll
  for (int i = 0; i < 8; ++i) {
    const int c = i * 256 + threadIdx.x;
    r1[i] = p1[c]; r2[i] = p2[c];
    m1 = fmaxf(m1, r1[i]); m2 = fmaxf(m2, r2[i]);
  }
  m1 = waveMax(m1); m2 = waveMax(m2);
  __shared__ float shA[8], shB[8];
  const int w = threadIdx.x >> 5;
  if ((threadIdx.x & 31) == 0) { shA[w] = m1; shB[w] = m2; }
  __syncthreads();
  float M1 = shA[0], M2 = shB[0];
#pragma unroll
  for (int i = 1; i < 8; ++i) { M1 = fmaxf(M1, shA[i]); M2 = fmaxf(M2, shB[i]); }
  float z1 = 0.f, z2 = 0.f;
#pragma unroll
  for (int i = 0; i < 8; ++i) {
    r1[i] = expf(r1[i] - M1); z1 += r1[i];
    r2[i] = expf(r2[i] - M2); z2 += r2[i];
  }
  z1 = waveSum(z1); z2 = waveSum(z2);
  __syncthreads();
  if ((threadIdx.x & 31) == 0) { shA[w] = z1; shB[w] = z2; }
  __syncthreads();
  float Z1 = 0.f, Z2 = 0.f;
#pragma unroll
  for (int i = 0; i < 8; ++i) { Z1 += shA[i]; Z2 += shB[i]; }
  const float lam = *lamp;
  const float i1 = 1.0f / Z1, i2 = lam / Z2;
#pragma unroll
  for (int i = 0; i < 8; ++i) {
    const int c = i * 256 + threadIdx.x;
    a[(size_t)row * SEQ + c] = (_Float16)(r1[i] * i1 - r2[i] * i2);
  }
}

// ---------------------------------------------------------------------------
// RMSNorm over 128, * subln_w * (1-LAMBDA_INIT); scatter heads back to [B,S,D]
// One block (128 threads) per (b,h,s).
// ---------------------------------------------------------------------------
__global__ __launch_bounds__(128) void rms_combine_kernel(
    const float* __restrict__ obuf, const float* __restrict__ subw,
    _Float16* __restrict__ o16)
{
  const int idx = blockIdx.x;          // (b*NH + h)*SEQ + s
  const int s = idx & (SEQ - 1);
  const int bh = idx >> 11;
  const int h = bh & (NH - 1), b = bh >> 3;
  const float v = obuf[(size_t)idx * HD2 + threadIdx.x];
  float sq = waveSum(v * v);
  __shared__ float sh[4];
  if ((threadIdx.x & 31) == 0) sh[threadIdx.x >> 5] = sq;
  __syncthreads();
  const float tot = sh[0] + sh[1] + sh[2] + sh[3];
  const float inv = rsqrtf(tot * (1.0f / HD2) + 1e-5f);
  const float outv = v * inv * subw[threadIdx.x] * (1.0f - LAMBDA_INIT);
  o16[((size_t)(b * SEQ + s)) * DIM + (size_t)h * HD2 + threadIdx.x] = (_Float16)outv;
}

// ---------------------------------------------------------------------------
// Host orchestration
// ---------------------------------------------------------------------------
extern "C" void kernel_launch(void* const* d_in, const int* in_sizes, int n_in,
                              void* d_out, int out_size, void* d_ws, size_t ws_size,
                              hipStream_t stream)
{
  const float* x      = (const float*)d_in[0];
  const float* ln1_w  = (const float*)d_in[1];
  const float* ln1_b  = (const float*)d_in[2];
  const float* wq     = (const float*)d_in[3];
  const float* wk     = (const float*)d_in[4];
  const float* wv     = (const float*)d_in[5];
  const float* wo     = (const float*)d_in[6];
  const float* lq1    = (const float*)d_in[7];
  const float* lk1    = (const float*)d_in[8];
  const float* lq2    = (const float*)d_in[9];
  const float* lk2    = (const float*)d_in[10];
  const float* subw   = (const float*)d_in[11];
  const float* ln2_w  = (const float*)d_in[12];
  const float* ln2_b  = (const float*)d_in[13];
  const float* w_in   = (const float*)d_in[14];
  const float* w_out  = (const float*)d_in[15];
  float* yout = (float*)d_out;

  char* p = (char*)d_ws;
  auto alloc = [&](size_t bytes) -> void* {
    void* r = (void*)p;
    p += (bytes + 255) & ~(size_t)255;
    return r;
  };
  _Float16* wqT   = (_Float16*)alloc((size_t)DIM * DIM * 2);
  _Float16* wkT   = (_Float16*)alloc((size_t)DIM * DIM * 2);
  _Float16* wvT   = (_Float16*)alloc((size_t)DIM * DIM * 2);
  _Float16* woT   = (_Float16*)alloc((size_t)DIM * DIM * 2);
  _Float16* winT  = (_Float16*)alloc((size_t)2 * FFN * DIM * 2);
  _Float16* woutT = (_Float16*)alloc((size_t)DIM * FFN * 2);
  _Float16* h16   = (_Float16*)alloc((size_t)MROWS * DIM * 2);
  _Float16* q16   = (_Float16*)alloc((size_t)BSZ * 16 * SEQ * HDQ * 2);
  _Float16* k16   = (_Float16*)alloc((size_t)BSZ * 16 * SEQ * HDQ * 2);
  _Float16* vT16  = (_Float16*)alloc((size_t)BSZ * NH * HD2 * SEQ * 2);
  float*    s1    = (float*)   alloc((size_t)SEQ * SEQ * 4);
  float*    s2    = (float*)   alloc((size_t)SEQ * SEQ * 4);
  _Float16* a16   = (_Float16*)alloc((size_t)SEQ * SEQ * 2);
  float*    obuf  = (float*)   alloc((size_t)BSZ * NH * SEQ * HD2 * 4);
  _Float16* o16   = (_Float16*)alloc((size_t)MROWS * DIM * 2);
  float*    ybuf  = (float*)   alloc((size_t)MROWS * DIM * 4);
  _Float16* h216  = (_Float16*)alloc((size_t)MROWS * DIM * 2);
  float*    gbuf  = (float*)   alloc((size_t)MROWS * FFN * 4);
  _Float16* t16   = (_Float16*)alloc((size_t)MROWS * FFN * 2);
  float*    lam   = (float*)   alloc(256);

  // lambda scalar
  lam_kernel<<<1, 32, 0, stream>>>(lq1, lk1, lq2, lk2, lam);

  // weight cast+transpose: in [K,N] -> f16 [N,K]
  transpose_cast_kernel<<<dim3(DIM/32, DIM/32), 256, 0, stream>>>(wq, wqT, DIM, DIM);
  transpose_cast_kernel<<<dim3(DIM/32, DIM/32), 256, 0, stream>>>(wk, wkT, DIM, DIM);
  transpose_cast_kernel<<<dim3(DIM/32, DIM/32), 256, 0, stream>>>(wv, wvT, DIM, DIM);
  transpose_cast_kernel<<<dim3(DIM/32, DIM/32), 256, 0, stream>>>(wo, woT, DIM, DIM);
  transpose_cast_kernel<<<dim3(2*FFN/32, DIM/32), 256, 0, stream>>>(w_in, winT, DIM, 2*FFN);
  transpose_cast_kernel<<<dim3(DIM/32, FFN/32), 256, 0, stream>>>(w_out, woutT, FFN, DIM);

  // ---- attention block ----
  layernorm_kernel<<<MROWS, 256, 0, stream>>>(x, ln1_w, ln1_b, h16);

  // q,k head-major f16 [B,2H,S,64]; v transposed head-major f16 [B,H,128,S]
  gemm_wmma<EPI_QK><<<dim3(DIM/64, MROWS/128), 256, 0, stream>>>(
      h16, wqT, nullptr, q16, nullptr, DIM, DIM, DIM, 0);
  gemm_wmma<EPI_QK><<<dim3(DIM/64, MROWS/128), 256, 0, stream>>>(
      h16, wkT, nullptr, k16, nullptr, DIM, DIM, DIM, 0);
  gemm_wmma<EPI_V><<<dim3(DIM/64, MROWS/128), 256, 0, stream>>>(
      h16, wvT, nullptr, vT16, nullptr, DIM, DIM, DIM, 0);

  for (int b = 0; b < BSZ; ++b) {
    for (int h = 0; h < NH; ++h) {
      const _Float16* qa = q16 + (size_t)(b * 16 + 2 * h)     * SEQ * HDQ;
      const _Float16* ka = k16 + (size_t)(b * 16 + 2 * h)     * SEQ * HDQ;
      const _Float16* qb = q16 + (size_t)(b * 16 + 2 * h + 1) * SEQ * HDQ;
      const _Float16* kb = k16 + (size_t)(b * 16 + 2 * h + 1) * SEQ * HDQ;
      // scores (scaled, causal-masked)
      gemm_wmma<EPI_SCORES><<<dim3(SEQ/64, SEQ/128), 256, 0, stream>>>(
          qa, ka, s1, nullptr, nullptr, HDQ, HDQ, HDQ, SEQ);
      gemm_wmma<EPI_SCORES><<<dim3(SEQ/64, SEQ/128), 256, 0, stream>>>(
          qb, kb, s2, nullptr, nullptr, HDQ, HDQ, HDQ, SEQ);
      // a = softmax(s1) - lam*softmax(s2)  (f16)
      softdiff_kernel<<<SEQ, 256, 0, stream>>>(s1, s2, lam, a16);
      // o = a @ v   (BT = vT head slice [128, S])
      const _Float16* vT = vT16 + (size_t)(b * NH + h) * HD2 * SEQ;
      float* oh = obuf + (size_t)(b * NH + h) * SEQ * HD2;
      gemm_wmma<EPI_F32><<<dim3(HD2/64, SEQ/128), 256, 0, stream>>>(
          a16, vT, oh, nullptr, nullptr, SEQ, SEQ, SEQ, HD2);
    }
  }

  // RMSNorm per (b,h,s), scale, re-concat heads -> o16 [B*S, D]
  rms_combine_kernel<<<BSZ * NH * SEQ, 128, 0, stream>>>(obuf, subw, o16);

  // y = x + o @ wo
  gemm_wmma<EPI_RESID><<<dim3(DIM/64, MROWS/128), 256, 0, stream>>>(
      o16, woT, ybuf, nullptr, x, DIM, DIM, DIM, DIM);

  // ---- SwishGLU FFN ----
  layernorm_kernel<<<MROWS, 256, 0, stream>>>(ybuf, ln2_w, ln2_b, h216);

  // g = h2 @ w_in[:, :FFN]   (winT rows 0..FFN-1)
  gemm_wmma<EPI_F32><<<dim3(FFN/64, MROWS/128), 256, 0, stream>>>(
      h216, winT, gbuf, nullptr, nullptr, DIM, DIM, DIM, FFN);
  // t = silu(g) * (h2 @ w_in[:, FFN:])   (f16)
  gemm_wmma<EPI_SILU><<<dim3(FFN/64, MROWS/128), 256, 0, stream>>>(
      h216, winT + (size_t)FFN * DIM, nullptr, t16, gbuf, DIM, DIM, DIM, FFN);
  // out = y + t @ w_out
  gemm_wmma<EPI_RESID><<<dim3(DIM/64, MROWS/128), 256, 0, stream>>>(
      t16, woutT, yout, nullptr, ybuf, FFN, FFN, FFN, DIM);

  (void)in_sizes; (void)n_in; (void)out_size; (void)ws_size;
}
